// MixtureOfExperts_19567871001015
// MI455X (gfx1250) — matrix operand order repaired and checked by reading, not compiled, not run on hardware
//
#include <hip/hip_runtime.h>
#include <hip/hip_bf16.h>

// ---------------- problem constants (match reference) ----------------
constexpr int kE   = 8;
constexpr int kB   = 2;
constexpr int kS   = 4096;
constexpr int kD   = 1024;
constexpr int kH   = 4096;
constexpr int kCap = 2048;
constexpr int kT   = kB * kS;          // 8192 tokens
constexpr int kTile = 1024;            // tokens per scan tile
constexpr int kNT  = kT / kTile;       // 8 tiles

// ---------------- WMMA vector types (gfx1250, wave32) ----------------
typedef __attribute__((ext_vector_type(16))) __bf16         v16bf;
typedef __attribute__((ext_vector_type(16))) unsigned short v16u;
typedef __attribute__((ext_vector_type(8)))  float          v8f;
typedef __attribute__((ext_vector_type(4)))  int            v4i;

#define AS1 __attribute__((address_space(1)))
#define AS3 __attribute__((address_space(3)))

// Async global->LDS (CDNA5); guarded so we fall back cleanly if absent.
#if defined(__has_builtin)
#if __has_builtin(__builtin_amdgcn_global_load_async_to_lds_b128) && \
    __has_builtin(__builtin_amdgcn_s_wait_asynccnt)
#define MOE_ASYNC_LDS 1
#endif
#endif
#ifndef MOE_ASYNC_LDS
#define MOE_ASYNC_LDS 0
#endif

// ---------------- helpers ----------------
__device__ __forceinline__ float gelu_tanh(float x) {
    const float c = 0.7978845608028654f;  // sqrt(2/pi)
    float x3 = x * x * x;
    return 0.5f * x * (1.0f + tanhf(c * (x + 0.044715f * x3)));
}
__device__ __forceinline__ void storeOut(float* p, float v) { *p = v; }
__device__ __forceinline__ void storeOut(__hip_bfloat16* p, float v) { *p = __float2bfloat16(v); }

// =====================================================================
// 1) Gating: logits = x @ Wg ; softmax ; top-2 ; renormalize
// =====================================================================
__global__ __launch_bounds__(256)
void gate_kernel(const float* __restrict__ x, const float* __restrict__ Wg,
                 int* __restrict__ idxTop, float* __restrict__ wTop) {
    __shared__ float sWg[kD * kE];
    for (int i = threadIdx.x; i < kD * kE; i += blockDim.x) sWg[i] = Wg[i];
    __syncthreads();

    const int wave = threadIdx.x >> 5;
    const int lane = threadIdx.x & 31;
    const int token = blockIdx.x * (blockDim.x >> 5) + wave;
    if (token >= kT) return;

    float acc[kE];
#pragma unroll
    for (int e = 0; e < kE; ++e) acc[e] = 0.0f;

    const float* xr = x + (size_t)token * kD;
    for (int d = lane; d < kD; d += 32) {
        float xv = xr[d];
#pragma unroll
        for (int e = 0; e < kE; ++e) acc[e] += xv * sWg[d * kE + e];
    }
#pragma unroll
    for (int e = 0; e < kE; ++e) {
#pragma unroll
        for (int o = 16; o > 0; o >>= 1) acc[e] += __shfl_xor(acc[e], o, 32);
    }
    if (lane == 0) {
        float m = acc[0];
#pragma unroll
        for (int e = 1; e < kE; ++e) m = fmaxf(m, acc[e]);
        float p[kE]; float sum = 0.0f;
#pragma unroll
        for (int e = 0; e < kE; ++e) { p[e] = __expf(acc[e] - m); sum += p[e]; }
        float inv = 1.0f / sum;
#pragma unroll
        for (int e = 0; e < kE; ++e) p[e] *= inv;
        int e0 = 0; float v0 = p[0];
#pragma unroll
        for (int e = 1; e < kE; ++e) if (p[e] > v0) { v0 = p[e]; e0 = e; }
        int e1 = (e0 == 0) ? 1 : 0; float v1 = p[e1];
#pragma unroll
        for (int e = 0; e < kE; ++e) if (e != e0 && p[e] > v1) { v1 = p[e]; e1 = e; }
        float s = v0 + v1 + 1e-9f;
        idxTop[token * 2 + 0] = e0;
        idxTop[token * 2 + 1] = e1;
        wTop[token * 2 + 0] = v0 / s;
        wTop[token * 2 + 1] = v1 / s;
    }
}

// =====================================================================
// 2a) per-tile (rank, expert) counts
// =====================================================================
__global__ __launch_bounds__(1024)
void count_kernel(const int* __restrict__ idxTop, int* __restrict__ tileCnt) {
    __shared__ int c[2 * kE];
    if (threadIdx.x < 2 * kE) c[threadIdx.x] = 0;
    __syncthreads();
    const int t = blockIdx.x * kTile + threadIdx.x;
    atomicAdd(&c[0 * kE + idxTop[t * 2 + 0]], 1);
    atomicAdd(&c[1 * kE + idxTop[t * 2 + 1]], 1);
    __syncthreads();
    if (threadIdx.x < 2 * kE) tileCnt[blockIdx.x * 2 * kE + threadIdx.x] = c[threadIdx.x];
}

// =====================================================================
// 2b) tile scan: rank-0 bases; rank-1 bases offset by kept0=min(raw0,CAP)
// =====================================================================
__global__ __launch_bounds__(32)
void scan_kernel(const int* __restrict__ tileCnt, int* __restrict__ tileBase,
                 int* __restrict__ exCount) {
    const int e = threadIdx.x;
    if (e >= kE) return;
    int run = 0;
    for (int tile = 0; tile < kNT; ++tile) {
        tileBase[tile * 2 * kE + e] = run;
        run += tileCnt[tile * 2 * kE + e];
    }
    const int kept0 = min(run, kCap);
    int run1 = kept0;
    for (int tile = 0; tile < kNT; ++tile) {
        tileBase[tile * 2 * kE + kE + e] = run1;
        run1 += tileCnt[tile * 2 * kE + kE + e];
    }
    exCount[e] = min(run1, kCap);
}

// =====================================================================
// 2c) assignment: exact token-order positions via ballot + wave scans
// =====================================================================
__global__ __launch_bounds__(1024)
void assign_kernel(const int* __restrict__ idxTop, const int* __restrict__ tileBase,
                   int* __restrict__ tokenSlot, int* __restrict__ expertTok) {
    __shared__ int wcnt[2][32][kE];
    __shared__ int wbase[2][32][kE];
    const int t = blockIdx.x * kTile + threadIdx.x;
    const int wave = threadIdx.x >> 5;
    const int lane = threadIdx.x & 31;
    const int e0 = idxTop[t * 2 + 0];
    const int e1 = idxTop[t * 2 + 1];
    int r0 = 0, r1 = 0;
    const unsigned lt = (lane == 31) ? 0x7fffffffu : ((1u << lane) - 1u);
#pragma unroll
    for (int eo = 0; eo < kE; ++eo) {
        unsigned m0 = (unsigned)__ballot(e0 == eo);
        unsigned m1 = (unsigned)__ballot(e1 == eo);
        if (lane == 0) { wcnt[0][wave][eo] = __popc(m0); wcnt[1][wave][eo] = __popc(m1); }
        if (e0 == eo) r0 = __popc(m0 & lt);
        if (e1 == eo) r1 = __popc(m1 & lt);
    }
    __syncthreads();
    if (threadIdx.x < 2 * kE) {
        const int k = threadIdx.x >> 3;
        const int e = threadIdx.x & 7;
        int run = 0;
        for (int w = 0; w < 32; ++w) { wbase[k][w][e] = run; run += wcnt[k][w][e]; }
    }
    __syncthreads();
    const int pos0 = tileBase[blockIdx.x * 2 * kE + e0]      + wbase[0][wave][e0] + r0;
    const int pos1 = tileBase[blockIdx.x * 2 * kE + kE + e1] + wbase[1][wave][e1] + r1;
    if (pos0 < kCap) { tokenSlot[t * 2 + 0] = e0 * kCap + pos0; expertTok[e0 * kCap + pos0] = t; }
    else             { tokenSlot[t * 2 + 0] = -1; }
    if (pos1 < kCap) { tokenSlot[t * 2 + 1] = e1 * kCap + pos1; expertTok[e1 * kCap + pos1] = t; }
    else             { tokenSlot[t * 2 + 1] = -1; }
}

// =====================================================================
// 3a) weight fp32 -> bf16 + K-block swizzle:
//     dst[e][k>>5][n][k&31] = bf16(src[e][k][n])   (B tiles become
//     contiguous, fragment-ordered 8KB global regions)
// =====================================================================
__global__ __launch_bounds__(256)
void cvt_swz_kernel(const float* __restrict__ src, __hip_bfloat16* __restrict__ dst,
                    int Kdim, int Ndim) {
    const int n = blockIdx.x * blockDim.x + threadIdx.x;   // column
    const int k = blockIdx.y;                              // row
    const int e = blockIdx.z;
    const size_t KN = (size_t)Kdim * Ndim;
    const float v = src[(size_t)e * KN + (size_t)k * Ndim + n];
    dst[(size_t)e * KN + ((size_t)(k >> 5) * Ndim + n) * 32 + (k & 31)] = __float2bfloat16(v);
}

// =====================================================================
// 3b) gather tokens into Xe[E][CAP][D] (bf16 row-major, zero pad)
// =====================================================================
__global__ __launch_bounds__(256)
void gather_kernel(const float* __restrict__ x, const int* __restrict__ expertTok,
                   const int* __restrict__ exCount, __hip_bfloat16* __restrict__ Xe) {
    const int row = blockIdx.x;             // e*CAP + slot
    const int e = row / kCap, slot = row % kCap;
    __hip_bfloat16* dst = Xe + (size_t)row * kD;
    if (slot >= exCount[e]) {
        for (int d = threadIdx.x; d < kD; d += blockDim.x) dst[d] = __float2bfloat16(0.0f);
        return;
    }
    const float* src = x + (size_t)expertTok[row] * kD;
    for (int d = threadIdx.x; d < kD; d += blockDim.x) dst[d] = __float2bfloat16(src[d]);
}

// =====================================================================
// 4) bf16 WMMA GEMM with double-buffered (async) LDS staging
//    A: [CAP][Kdim] bf16 row-major; Wsw: [Kdim/32][Ndim][32] bf16
//    block tile 128x128, 8 waves -> 64x32 each = 4x2 wmma accumulators
// =====================================================================
template <int Kdim, int Ndim, bool GELU, typename OutT>
__global__ __launch_bounds__(256)
void gemm_kernel(const __hip_bfloat16* __restrict__ Abase,
                 const __hip_bfloat16* __restrict__ Wbase,
                 const float* __restrict__ biasBase,
                 OutT* __restrict__ Cbase) {
    constexpr int BM = 128, BN = 128, BK = 32;
    constexpr int SAS = BK + 8;   // 40 shorts = 80B row stride (16B aligned, conflict-free)
    constexpr int SBS = BK + 8;   // sB is [col][k] fragment-ordered

    const int e = blockIdx.z;
    const unsigned short* A = (const unsigned short*)(Abase + (size_t)e * kCap * Kdim);
    const unsigned short* W = (const unsigned short*)(Wbase + (size_t)e * Kdim * Ndim);
    const float* bias = biasBase + (size_t)e * Ndim;
    OutT* C = Cbase + (size_t)e * kCap * Ndim;

    __shared__ unsigned short sA[2][BM * SAS];   // 2 x 10KB
    __shared__ unsigned short sB[2][BN * SBS];   // 2 x 10KB

    const int tid = threadIdx.x;
    const int wave = tid >> 5, lane = tid & 31;
    const int waveM = wave >> 2, waveN = wave & 3;          // 2 x 4 wave grid
    const int laneLo = lane & 15, laneHi = lane >> 4;

    const int blockM = blockIdx.y * BM;
    const int blockN = blockIdx.x * BN;

    // staging coords: each thread moves 32B of A and 32B of B per k-step
    const int arow = tid >> 1, ahalf = tid & 1;             // A: [row][khalf*16..]
    const int bcol = tid >> 1, bhalf = tid & 1;             // B: [col][khalf*16..]

    auto stage = [&](int buf, int k0) {
        const unsigned short* ga = A + (size_t)(blockM + arow) * Kdim + k0 + ahalf * 16;
        unsigned short*       la = &sA[buf][arow * SAS + ahalf * 16];
        // swizzled W: tile base is contiguous [col][32]
        const unsigned short* gb = W + ((size_t)(k0 >> 5) * Ndim + blockN + bcol) * 32 + bhalf * 16;
        unsigned short*       lb = &sB[buf][bcol * SBS + bhalf * 16];
#if MOE_ASYNC_LDS
        __builtin_amdgcn_global_load_async_to_lds_b128((AS1 v4i*)ga, (AS3 v4i*)la, 0, 0);
        __builtin_amdgcn_global_load_async_to_lds_b128((AS1 v4i*)(ga + 8), (AS3 v4i*)(la + 8), 0, 0);
        __builtin_amdgcn_global_load_async_to_lds_b128((AS1 v4i*)gb, (AS3 v4i*)lb, 0, 0);
        __builtin_amdgcn_global_load_async_to_lds_b128((AS1 v4i*)(gb + 8), (AS3 v4i*)(lb + 8), 0, 0);
#else
        const uint4* ga4 = (const uint4*)ga;
        const uint4* gb4 = (const uint4*)gb;
        uint4 a0 = ga4[0], a1 = ga4[1];
        uint4 b0 = gb4[0], b1 = gb4[1];
        *(uint4*)la       = a0; *(uint4*)(la + 8) = a1;
        *(uint4*)lb       = b0; *(uint4*)(lb + 8) = b1;
#endif
    };
    auto stage_wait = [&]() {
#if MOE_ASYNC_LDS
        __builtin_amdgcn_s_wait_asynccnt(0);
#endif
        __syncthreads();
    };

    v8f acc[4][2];
#pragma unroll
    for (int mi = 0; mi < 4; ++mi)
#pragma unroll
        for (int ni = 0; ni < 2; ++ni) acc[mi][ni] = (v8f)(0.0f);

    stage(0, 0);
    stage_wait();

    for (int k0 = 0; k0 < Kdim; k0 += BK) {
        const int cur = (k0 / BK) & 1;
        if (k0 + BK < Kdim) stage(cur ^ 1, k0 + BK);  // overlap next-tile copy with WMMA

        // fragments per CDNA5 16-bit WMMA lane layouts (all reads 2x16B contiguous)
        v16u aU[4];
#pragma unroll
        for (int mi = 0; mi < 4; ++mi) {
            const unsigned short* rp = &sA[cur][(waveM * 64 + mi * 16 + laneLo) * SAS];
#pragma unroll
            for (int j = 0; j < 8; ++j) {
                const int kk = (j < 4) ? (laneHi * 8 + 2 * j) : (16 + laneHi * 8 + 2 * (j - 4));
                aU[mi][2 * j]     = rp[kk];
                aU[mi][2 * j + 1] = rp[kk + 1];
            }
        }
        v16u bU[2];
#pragma unroll
        for (int ni = 0; ni < 2; ++ni) {
            const unsigned short* cp =
                &sB[cur][(waveN * 32 + ni * 16 + laneLo) * SBS + laneHi * 8];
#pragma unroll
            for (int j = 0; j < 16; ++j) bU[ni][j] = cp[(j < 8) ? j : (j + 8)];
        }
#pragma unroll
        for (int mi = 0; mi < 4; ++mi) {
            v16bf av = __builtin_bit_cast(v16bf, aU[mi]);
#pragma unroll
            for (int ni = 0; ni < 2; ++ni) {
                v16bf bv = __builtin_bit_cast(v16bf, bU[ni]);
                acc[mi][ni] = __builtin_amdgcn_wmma_f32_16x16x32_bf16(
                    false, av, false, bv, (short)0, acc[mi][ni], false, false);
            }
        }
        stage_wait();
    }

    // epilogue: acc VGPR r -> (M = base + r + 8*laneHi, N = base + laneLo)
#pragma unroll
    for (int mi = 0; mi < 4; ++mi) {
#pragma unroll
        for (int ni = 0; ni < 2; ++ni) {
            const int col = blockN + waveN * 32 + ni * 16 + laneLo;
            const float bv = bias[col];
#pragma unroll
            for (int r = 0; r < 8; ++r) {
                const int rowg = blockM + waveM * 64 + mi * 16 + r + laneHi * 8;
                float v = acc[mi][ni][r] + bv;
                if (GELU) v = gelu_tanh(v);
                storeOut(&C[(size_t)rowg * Ndim + col], v);
            }
        }
    }
}

// =====================================================================
// 5) combine: out[t] = sum_k w[t,k] * Ye[slot(t,k)]
// =====================================================================
__global__ __launch_bounds__(256)
void combine_kernel(const int* __restrict__ tokenSlot, const float* __restrict__ wTop,
                    const float* __restrict__ Ye, float* __restrict__ out) {
    const int t = blockIdx.x;
    const int s0 = tokenSlot[t * 2 + 0];
    const int s1 = tokenSlot[t * 2 + 1];
    const float w0 = wTop[t * 2 + 0];
    const float w1 = wTop[t * 2 + 1];
    const float* y0 = Ye + (size_t)(s0 < 0 ? 0 : s0) * kD;
    const float* y1 = Ye + (size_t)(s1 < 0 ? 0 : s1) * kD;
    float* o = out + (size_t)t * kD;
    for (int d = threadIdx.x; d < kD; d += blockDim.x) {
        float v = 0.0f;
        if (s0 >= 0) v += w0 * y0[d];
        if (s1 >= 0) v += w1 * y1[d];
        o[d] = v;
    }
}

// =====================================================================
// launch
// =====================================================================
extern "C" void kernel_launch(void* const* d_in, const int* in_sizes, int n_in,
                              void* d_out, int out_size, void* d_ws, size_t ws_size,
                              hipStream_t stream) {
    const float* x  = (const float*)d_in[0];   // [B,S,D]
    const float* Wg = (const float*)d_in[1];   // [D,E]
    const float* W1 = (const float*)d_in[2];   // [E,D,H]
    const float* b1 = (const float*)d_in[3];   // [E,H]
    const float* W2 = (const float*)d_in[4];   // [E,H,D]
    const float* b2 = (const float*)d_in[5];   // [E,D]
    float* out = (float*)d_out;

    char* ws = (char*)d_ws;
    size_t off = 0;
    auto take = [&](size_t bytes) -> void* {
        void* p = ws + off;
        off += (bytes + 255) & ~(size_t)255;
        return p;
    };
    int*   idxTop    = (int*)  take((size_t)kT * 2 * sizeof(int));
    float* wTop      = (float*)take((size_t)kT * 2 * sizeof(float));
    int*   tileCnt   = (int*)  take((size_t)kNT * 2 * kE * sizeof(int));
    int*   tileBase  = (int*)  take((size_t)kNT * 2 * kE * sizeof(int));
    int*   exCount   = (int*)  take((size_t)kE * sizeof(int));
    int*   tokenSlot = (int*)  take((size_t)kT * 2 * sizeof(int));
    int*   expertTok = (int*)  take((size_t)kE * kCap * sizeof(int));
    __hip_bfloat16* W1b = (__hip_bfloat16*)take((size_t)kE * kD * kH * sizeof(__hip_bfloat16));
    __hip_bfloat16* W2b = (__hip_bfloat16*)take((size_t)kE * kH * kD * sizeof(__hip_bfloat16));
    __hip_bfloat16* Xe  = (__hip_bfloat16*)take((size_t)kE * kCap * kD * sizeof(__hip_bfloat16));
    __hip_bfloat16* He  = (__hip_bfloat16*)take((size_t)kE * kCap * kH * sizeof(__hip_bfloat16));
    float*          Ye  = (float*)         take((size_t)kE * kCap * kD * sizeof(float));

    // 1) gating
    gate_kernel<<<kT / 8, 256, 0, stream>>>(x, Wg, idxTop, wTop);
    // 2) capacity assignment
    count_kernel<<<kNT, kTile, 0, stream>>>(idxTop, tileCnt);
    scan_kernel<<<1, 32, 0, stream>>>(tileCnt, tileBase, exCount);
    assign_kernel<<<kNT, kTile, 0, stream>>>(idxTop, tileBase, tokenSlot, expertTok);
    // 3) precision conversion (K-block swizzled weights) + dispatch
    {
        dim3 gw1(kH / 256, kD, kE);
        cvt_swz_kernel<<<gw1, 256, 0, stream>>>(W1, W1b, kD, kH);
        dim3 gw2(kD / 256, kH, kE);
        cvt_swz_kernel<<<gw2, 256, 0, stream>>>(W2, W2b, kH, kD);
    }
    gather_kernel<<<kE * kCap, 256, 0, stream>>>(x, expertTok, exCount, Xe);
    // 4) expert FFN GEMMs (bf16 WMMA, async double-buffered staging)
    {
        dim3 g1(kH / 128, kCap / 128, kE);
        gemm_kernel<kD, kH, true, __hip_bfloat16><<<g1, 256, 0, stream>>>(Xe, W1b, b1, He);
        dim3 g2(kD / 128, kCap / 128, kE);
        gemm_kernel<kH, kD, false, float><<<g2, 256, 0, stream>>>(He, W2b, b2, Ye);
    }
    // 5) combine
    combine_kernel<<<kT, 256, 0, stream>>>(tokenSlot, wTop, Ye, out);
    (void)in_sizes; (void)n_in; (void)out_size; (void)ws_size;
}
